// HungarianMatcher_34162169872919
// MI455X (gfx1250) — matrix operand order repaired and checked by reading, not compiled, not run on hardware
//
#include <hip/hip_runtime.h>
#include <math.h>

// ---------------------------------------------------------------------------
// HungarianMatcher for MI455X (gfx1250, wave32).
//   bs=32, Q=512, classes=92 (pad 96), T=24 (pad 32), N=768.
// cost_class is computed as probs_f16 @ onehot_f16 via v_wmma_f32_16x16x32_f16.
// The 768-step assignment scan runs as ONE persistent workgroup keeping the
// per-column top-2 state in LDS and updating it incrementally (only columns
// {col, 0, 1} change per step), so each iteration is O(Q+N) instead of O(Q*N).
// ---------------------------------------------------------------------------

typedef __attribute__((ext_vector_type(16))) _Float16 v16h;
typedef __attribute__((ext_vector_type(8)))  _Float16 v8h;
typedef __attribute__((ext_vector_type(8)))  float    v8f;

namespace {
constexpr int BS  = 32;
constexpr int Q   = 512;
constexpr int NC  = 92;
constexpr int NCP = 96;   // K padded to multiple of 32
constexpr int T   = 24;
constexpr int TP  = 32;   // N-tile padded to multiple of 16
constexpr int NT  = BS * T;   // 768
constexpr float SQRT2   = 1.41421356237309515f;
constexpr float INV_SS2 = 0.28284271247461906f;   // 1/(2.5*sqrt(2))
}

// --------------------------- K1: softmax -> f16 ----------------------------
// One wave per row of [16384 x 92]; output padded to 96 halves.
__global__ __launch_bounds__(128) void softmax_f16_kernel(
    const float* __restrict__ logits, _Float16* __restrict__ probs) {
  const int lane = threadIdx.x & 31;
  const int row  = blockIdx.x * 4 + (threadIdx.x >> 5);
  const float* src = logits + row * NC;
  float v0 = src[lane];
  float v1 = src[lane + 32];
  float v2 = (lane < NC - 64) ? src[lane + 64] : -3.4e38f;
  float m = fmaxf(v0, fmaxf(v1, v2));
#pragma unroll
  for (int off = 16; off; off >>= 1) m = fmaxf(m, __shfl_xor(m, off, 32));
  float e0 = expf(v0 - m);
  float e1 = expf(v1 - m);
  float e2 = (lane < NC - 64) ? expf(v2 - m) : 0.0f;
  float s = e0 + e1 + e2;
#pragma unroll
  for (int off = 16; off; off >>= 1) s += __shfl_xor(s, off, 32);
  float inv = 1.0f / s;
  _Float16* dst = probs + row * NCP;
  dst[lane]      = (_Float16)(e0 * inv);
  dst[lane + 32] = (_Float16)(e1 * inv);
  dst[lane + 64] = (_Float16)(e2 * inv);   // lanes 28..31 write the 0-pad
}

// --------------------------- K2: one-hot B matrix --------------------------
// Layout [b][t(32)][k(96)] f16: per-column K is contiguous for WMMA B loads.
__global__ __launch_bounds__(256) void onehot_kernel(
    const int* __restrict__ labels, _Float16* __restrict__ onehot) {
  int idx = blockIdx.x * 256 + threadIdx.x;
  if (idx >= BS * TP * NCP) return;
  int k = idx % NCP;
  int t = (idx / NCP) % TP;
  int b = idx / (NCP * TP);
  float v = 0.0f;
  if (t < T && k == labels[b * T + t]) v = 1.0f;
  onehot[idx] = (_Float16)v;
}

// ---------------- K3: cost_class = -(probs @ onehot) via WMMA --------------
// Per batch: [512x96]f16 @ [96x24(pad32)]f16 -> f32. One wave per 16x16 tile,
// 3 accumulating v_wmma_f32_16x16x32_f16 along K.
__global__ __launch_bounds__(256) void class_gemm_kernel(
    const _Float16* __restrict__ probs, const _Float16* __restrict__ onehot,
    float* __restrict__ ccls) {
  const int lane = threadIdx.x & 31;
  const int gw   = blockIdx.x * 8 + (threadIdx.x >> 5);  // 0..2047
  const int b    = gw >> 6;
  const int rem  = gw & 63;
  const int mt   = rem >> 1;       // 32 M-tiles of 16 rows
  const int nt   = rem & 1;        // 2 N-tiles of 16 cols
  const int hi   = lane >> 4;      // lane half
  const int l16  = lane & 15;

  // A fragment source: row q, K contiguous lo:{0-15} hi:{16-31}
  const int q = mt * 16 + l16;
  const _Float16* arow = probs + (b * Q + q) * NCP;
  // B fragment source: col t, K contiguous lo:{0-15} hi:{16-31}
  const int t = nt * 16 + l16;
  const _Float16* brow = onehot + (b * TP + t) * NCP;

  union AFrag { v16h v; _Float16 h[16]; } a;
  union CFrag { v8f  v; float    f[8];  } c;
  c.v = (v8f){};
#pragma unroll
  for (int ks = 0; ks < 3; ++ks) {
    const int k0 = ks * 32;
    const v8h a0 = *(const v8h*)(arow + k0 + (hi ? 16 : 0));
    const v8h a1 = *(const v8h*)(arow + k0 + (hi ? 24 : 8));
#pragma unroll
    for (int e = 0; e < 8; ++e) { a.h[e] = a0[e]; a.h[e + 8] = a1[e]; }
    const v16h bm = *(const v16h*)(brow + k0 + (hi ? 16 : 0));
    c.v = __builtin_amdgcn_wmma_f32_16x16x32_f16(
        /*neg_a=*/false, a.v, /*neg_b=*/false, bm,
        /*c_mod=*/(short)0, c.v, /*reuse_a=*/false, /*reuse_b=*/false);
  }
  // C/D: VGPR r -> M = r + 8*hi, N = lane%16. Store column-major [n][q].
  const int tout = nt * 16 + l16;
  if (tout < T) {
    const int n = b * T + tout;
#pragma unroll
    for (int r = 0; r < 8; ++r) {
      const int qo = mt * 16 + r + hi * 8;
      ccls[n * Q + qo] = -c.f[r];
    }
  }
}

// ------------- K4: bbox-L1 + GIoU + class -> fx (B matrix, [n][q]) ---------
__global__ __launch_bounds__(256) void fx_kernel(
    const float* __restrict__ pred_boxes, const float* __restrict__ tgt_boxes,
    const float* __restrict__ ccls, float* __restrict__ Bm) {
  int idx = blockIdx.x * 256 + threadIdx.x;
  if (idx >= NT * Q) return;
  int n = idx / Q, q = idx - n * Q, b = n / T;
  const float* pb = pred_boxes + (b * Q + q) * 4;
  const float* tb = tgt_boxes + n * 4;
  float cx = pb[0], cy = pb[1], w = pb[2], h = pb[3];
  float tx = tb[0], ty = tb[1], tw = tb[2], th = tb[3];
  float cbbox = fabsf(cx - tx) + fabsf(cy - ty) + fabsf(w - tw) + fabsf(h - th);
  // cxcywh -> xyxy
  float ax1 = cx - 0.5f * w,  ay1 = cy - 0.5f * h;
  float ax2 = cx + 0.5f * w,  ay2 = cy + 0.5f * h;
  float bx1 = tx - 0.5f * tw, by1 = ty - 0.5f * th;
  float bx2 = tx + 0.5f * tw, by2 = ty + 0.5f * th;
  float area1 = (ax2 - ax1) * (ay2 - ay1);
  float area2 = (bx2 - bx1) * (by2 - by1);
  float iw = fmaxf(fminf(ax2, bx2) - fmaxf(ax1, bx1), 0.0f);
  float ih = fmaxf(fminf(ay2, by2) - fmaxf(ay1, by1), 0.0f);
  float inter = iw * ih;
  float uni = area1 + area2 - inter;
  float iou = inter / uni;
  float ew = fmaxf(fmaxf(ax2, bx2) - fminf(ax1, bx1), 0.0f);
  float eh = fmaxf(fmaxf(ay2, by2) - fminf(ay1, by1), 0.0f);
  float enc = ew * eh;
  float giou = iou - (enc - uni) / enc;
  float C = cbbox + ccls[idx] - giou;
  // prob = Phi((-C - mu)/sigma) with mu=-5.5, sigma=2.5; fx = 0.3*sqrt2*erfinv
  float prob = 0.5f * (1.0f + erff((5.5f - C) * INV_SS2));
  Bm[idx] = 0.3f * SQRT2 * erfinvf(prob);   // HalfNormal icdf: always >= 0
}

// ------------------------------- K5: scan ----------------------------------
__device__ inline void argmax_combine(float& v, int& i, float ov, int oi) {
  if (ov > v || (ov == v && oi < i)) { v = ov; i = oi; }
}
__device__ inline void wave_argmax(float& v, int& i) {
#pragma unroll
  for (int off = 16; off; off >>= 1) {
    float ov = __shfl_xor(v, off, 32);
    int   oi = __shfl_xor(i, off, 32);
    argmax_combine(v, i, ov, oi);
  }
}
__device__ inline void wave_top2(float& a, float& b) {
#pragma unroll
  for (int off = 16; off; off >>= 1) {
    float oa = __shfl_xor(a, off, 32);
    float ob = __shfl_xor(b, off, 32);
    if (oa > a) { b = fmaxf(a, ob); a = oa; }
    else        { b = fmaxf(b, oa); }
  }
}
__device__ inline void column_top2(const float* __restrict__ Bm, int c,
                                   int lane, float* t1, float* t2) {
  float a = -3.4e38f, b = -3.4e38f;
  for (int q = lane; q < Q; q += 32) {
    float v = fminf(Bm[c * Q + q], 100.0f);
    if (v > a) { b = a; a = v; } else if (v > b) b = v;
  }
  wave_top2(a, b);
  if (lane == 0) { t1[c] = a; t2[c] = b; }
}

__global__ __launch_bounds__(512) void assign_scan_kernel(float* __restrict__ Bm) {
  __shared__ float s_t1[NT], s_t2[NT];
  __shared__ float s_rv[16];
  __shared__ int   s_ri[16];
  __shared__ int   s_col, s_row;
  const int tid = threadIdx.x, lane = tid & 31, wid = tid >> 5;

  // initial per-column top-2 of min(B, 100)
  for (int c = wid; c < NT; c += 16) column_top2(Bm, c, lane, s_t1, s_t2);
  __syncthreads();

  for (int it = 0; it < NT; ++it) {
    // (1) col = argmax_c (top1 - top2) == argmax |diff(top2)|, first-idx ties
    float v = -1.0f; int idx = 0x7fffffff;
    for (int c = tid; c < NT; c += 512)
      argmax_combine(v, idx, s_t1[c] - s_t2[c], c);
    wave_argmax(v, idx);
    if (lane == 0) { s_rv[wid] = v; s_ri[wid] = idx; }
    __syncthreads();
    if (wid == 0) {
      float vv = (lane < 16) ? s_rv[lane] : -3.4e38f;
      int   ii = (lane < 16) ? s_ri[lane] : 0x7fffffff;
      wave_argmax(vv, ii);
      if (lane == 0) s_col = ii;
    }
    __syncthreads();
    const int col = s_col;

    // (2) row = argmax_q B[q, col] on raw values, first-idx ties
    float bv = Bm[col * Q + tid];
    int   ri = tid;
    wave_argmax(bv, ri);
    if (lane == 0) { s_rv[wid] = bv; s_ri[wid] = ri; }
    __syncthreads();
    if (wid == 0) {
      float vv = (lane < 16) ? s_rv[lane] : -3.4e38f;
      int   ii = (lane < 16) ? s_ri[lane] : 0x7fffffff;
      wave_argmax(vv, ii);
      if (lane == 0) s_row = ii;
    }
    __syncthreads();
    const int row = s_row;

    // (3) updates in reference order: B[:,col]=0; B[row,{0,1}]=0; B[row,col]=-1e-7
    Bm[col * Q + tid] = 0.0f;
    __threadfence();
    __syncthreads();
    if (tid == 0) {
      Bm[0 * Q + row]   = 0.0f;
      Bm[1 * Q + row]   = 0.0f;
      Bm[col * Q + row] = -1e-7f;
      s_t1[col] = 0.0f;           // column col is now {0,...,0,-1e-7}
      s_t2[col] = 0.0f;
      __threadfence();
    }
    __syncthreads();

    // (4) only columns 0 and 1 need refreshing (col handled analytically)
    if (wid < 2 && wid != col) column_top2(Bm, wid, lane, s_t1, s_t2);
    __syncthreads();
  }
}

// --------------------------- K6: emit (B < 0) as [Q,N] ---------------------
__global__ __launch_bounds__(256) void emit_kernel(
    const float* __restrict__ Bm, float* __restrict__ out) {
  int idx = blockIdx.x * 256 + threadIdx.x;
  if (idx >= NT * Q) return;
  int n = idx / Q, q = idx - n * Q;
  out[q * NT + n] = (Bm[idx] < 0.0f) ? 1.0f : 0.0f;
}

// ---------------------------------------------------------------------------
extern "C" void kernel_launch(void* const* d_in, const int* in_sizes, int n_in,
                              void* d_out, int out_size, void* d_ws, size_t ws_size,
                              hipStream_t stream) {
  const float* pred_logits = (const float*)d_in[0];   // [32,512,92]
  const float* pred_boxes  = (const float*)d_in[1];   // [32,512,4]
  const float* tgt_boxes   = (const float*)d_in[2];   // [768,4]
  const int*   tgt_labels  = (const int*)d_in[3];     // [768]
  (void)in_sizes; (void)n_in; (void)out_size; (void)ws_size;

  char* ws = (char*)d_ws;
  _Float16* probs  = (_Float16*)(ws);                          // 16384*96*2   = 3,145,728 B
  _Float16* onehot = (_Float16*)(ws + 3145728);                // 32*32*96*2   =   196,608 B
  float*    ccls   = (float*)(ws + 3145728 + 196608);          // 768*512*4    = 1,572,864 B
  float*    Bm     = (float*)(ws + 3145728 + 196608 + 1572864);// 768*512*4    = 1,572,864 B

  softmax_f16_kernel<<<4096, 128, 0, stream>>>(pred_logits, probs);
  onehot_kernel<<<(BS * TP * NCP + 255) / 256, 256, 0, stream>>>(tgt_labels, onehot);
  class_gemm_kernel<<<256, 256, 0, stream>>>(probs, onehot, ccls);
  fx_kernel<<<(NT * Q + 255) / 256, 256, 0, stream>>>(pred_boxes, tgt_boxes, ccls, Bm);
  assign_scan_kernel<<<1, 512, 0, stream>>>(Bm);
  emit_kernel<<<(NT * Q + 255) / 256, 256, 0, stream>>>(Bm, (float*)d_out);
}
